// AttentionActorCritic_60258391163472
// MI455X (gfx1250) — matrix operand order repaired and checked by reading, not compiled, not run on hardware
//
#include <hip/hip_runtime.h>
#include <hip/hip_bf16.h>

typedef __attribute__((ext_vector_type(16))) _Float16 v16h;
typedef __attribute__((ext_vector_type(8)))  float    v8f;

#define OUT_COLS 268
#define NEG_INF_F (-1e30f)

#if defined(__has_builtin)
# if __has_builtin(__builtin_amdgcn_tensor_load_to_lds) && __has_builtin(__builtin_amdgcn_s_wait_tensorcnt)
#  define HAVE_TDM 1
# endif
#endif
#ifndef HAVE_TDM
# define HAVE_TDM 0
#endif

struct Params {
  const float *self_feat, *entities;
  const int   *mask;
  const float *hx, *cx;
  const float *W_se,*b_se,*W_ee,*b_ee,*W_q,*b_q,*W_k,*b_k,*W_v,*b_v;
  const float *W_p1,*b_p1,*W_p2,*b_p2,*W_ih,*b_ih,*W_hh,*b_hh;
  const float *W_a,*b_a,*W_mx,*b_mx,*W_my,*b_my,*W_fi,*b_fi,*W_he,*b_he,*W_ra,*b_ra,*W_rb,*b_rb;
  float* out;
};

__device__ __forceinline__ v8f wmma16(v16h a, v16h b, v8f c) {
  return __builtin_amdgcn_wmma_f32_16x16x32_f16(false, a, false, b, (short)0, c, false, false);
}

// B fragment (32x16 f16) from global row-major f32 W[K][NC]; rows >= kmax read as 0.
__device__ __forceinline__ v16h load_bfrag_g(const float* W, int NC, int kb, int cb,
                                             int kmax, int lm, int hi) {
  v16h b;
#pragma unroll
  for (int j = 0; j < 8; ++j) {
    int k0 = kb + (j & 3) * 2 + hi * 8 + (j >> 2) * 16;
    float x0 = (k0     < kmax) ? W[(long long)k0    *NC + cb + lm] : 0.f;
    float x1 = (k0 + 1 < kmax) ? W[(long long)(k0+1)*NC + cb + lm] : 0.f;
    b[2*j]   = (_Float16)x0;
    b[2*j+1] = (_Float16)x1;
  }
  return b;
}

// A fragment (16x32 f16) from LDS f16 row-major S[rows][RC], row base rb, k base kb.
__device__ __forceinline__ v16h load_afrag_lds(const _Float16* S, int RC, int rb, int kb,
                                               int lm, int hi) {
  v16h a;
#pragma unroll
  for (int j = 0; j < 8; ++j) {
    int k = kb + ((j & 3) * 2 + hi * 8 + (j >> 2) * 16);
    a[2*j]   = S[(rb + lm) * RC + k];
    a[2*j+1] = S[(rb + lm) * RC + k + 1];
  }
  return a;
}

// A fragment (16x32 f16) from LDS f32 row-major; k >= kmax -> 0.
__device__ __forceinline__ v16h load_afrag_lds_f32(const float* S, int RC, int rb,
                                                   int kmax, int lm, int hi) {
  v16h a;
#pragma unroll
  for (int j = 0; j < 8; ++j) {
    int k = (j & 3) * 2 + hi * 8 + (j >> 2) * 16;
    float x0 = (k     < kmax) ? S[(rb + lm) * RC + k]     : 0.f;
    float x1 = (k + 1 < kmax) ? S[(rb + lm) * RC + k + 1] : 0.f;
    a[2*j]   = (_Float16)x0;
    a[2*j+1] = (_Float16)x1;
  }
  return a;
}

// A fragment (16x32 f16) from global f32, row stride rs, base row rowbase; k >= kmax -> 0.
__device__ __forceinline__ v16h load_afrag_g(const float* X, int rs, long long rowbase,
                                             int kmax, int lm, int hi) {
  v16h a;
  const float* p = X + (rowbase + lm) * (long long)rs;
#pragma unroll
  for (int j = 0; j < 8; ++j) {
    int k = (j & 3) * 2 + hi * 8 + (j >> 2) * 16;
    a[2*j]   = (_Float16)((k     < kmax) ? p[k]     : 0.f);
    a[2*j+1] = (_Float16)((k + 1 < kmax) ? p[k + 1] : 0.f);
  }
  return a;
}

__device__ __forceinline__ float sigm(float x) { return 1.f / (1.f + __expf(-x)); }

#if HAVE_TDM
typedef __attribute__((ext_vector_type(4))) unsigned int tdm_v4u;
typedef __attribute__((ext_vector_type(8))) int          tdm_v8i;
typedef __attribute__((ext_vector_type(4))) int          tdm_v4i;

// 1-D TDM copy: ndw dwords from global gsrc into LDS at byte offset lds_off.
// D# per CDNA5 ISA ch.8: group0 = {count=1, lds_addr, global_addr, type=2},
// group1 = {data_size=4B, tensor_dim0=tile_dim0=ndw, tensor_dim1=tile_dim1=1}.
__device__ __forceinline__ void tdm_load_1d(const float* gsrc, unsigned lds_off, int ndw) {
  unsigned long long ga = (unsigned long long)(const void*)gsrc;
  tdm_v4u g0;
  g0.x = 1u;                                               // count=1, user mode
  g0.y = lds_off;                                          // lds_addr
  g0.z = (unsigned)(ga & 0xffffffffu);                     // global_addr[31:0]
  g0.w = (unsigned)((ga >> 32) & 0x1ffffffu) | (2u << 30); // global_addr[56:32] | type=2
  tdm_v8i g1;
  g1[0] = 0x20000;                                 // data_size=2 (4 bytes)
  g1[1] = (int)(((unsigned)ndw & 0xffffu) << 16);  // tensor_dim0[15:0] @ bits 63:48
  g1[2] = (int)((((unsigned)ndw >> 16) & 0xffffu) | (1u << 16)); // dim0[31:16], tensor_dim1=1
  g1[3] = (int)(((unsigned)ndw & 0xffffu) << 16);  // tile_dim0 @ bits 127:112
  g1[4] = 1;                                       // tile_dim1=1, tile_dim2=0
  g1[5] = ndw;                                     // tensor_dim0_stride[31:0]
  g1[6] = 0;
  g1[7] = 0;
  tdm_v4i gz = {0, 0, 0, 0};
#if __clang_major__ >= 23
  tdm_v8i gz8 = {0, 0, 0, 0, 0, 0, 0, 0};
  __builtin_amdgcn_tensor_load_to_lds(g0, g1, gz, gz, gz8, 0);
#else
  __builtin_amdgcn_tensor_load_to_lds(g0, g1, gz, gz, 0);
#endif
}
#endif

__global__ __launch_bounds__(128)
void aac_fused_kernel(Params p) {
  const int tid  = threadIdx.x;
  const int lane = tid & 31;
  const int wv   = tid >> 5;       // wave 0..3 (== attention head)
  const int lm   = lane & 15;      // A-row within tile / C-D column
  const int hi   = lane >> 4;      // lane half
  const int b0   = blockIdx.x * 16;
  const int cb   = wv * 16;        // this wave's 16-col slice of 64-wide outputs

  __shared__ _Float16 comb_h[16][128];    // [self_embed(0..63) | attn_out(64..127)]
  __shared__ float    q_s[16][64];
  __shared__ _Float16 ent_h[2][128][64];  // double-buffered entity embeddings
  __shared__ _Float16 r1_h[16][128];
  __shared__ _Float16 react_h[16][128];
  __shared__ _Float16 hx_h[16][128];
  __shared__ _Float16 feat_h[16][128];
  __shared__ _Float16 hh_h[16][128];
  __shared__ unsigned int mbits[16][4];
#if HAVE_TDM
  __shared__ float fbuf[2][128 * 12];     // double-buffered entity features (TDM dest)
#endif

  // ---- stage entity mask as bitmask + hx as f16 ----
  for (int t = tid; t < 64; t += 128) {
    int b = t >> 2, wi = t & 3;
    unsigned int bits = 0;
    for (int i = 0; i < 32; ++i)
      bits |= (p.mask[(long long)(b0 + b) * 128 + wi * 32 + i] != 0 ? 1u : 0u) << i;
    mbits[b][wi] = bits;
  }
  for (int t = tid; t < 16 * 128; t += 128) {
    int b = t >> 7, j = t & 127;
    hx_h[b][j] = (_Float16)p.hx[(long long)(b0 + b) * 128 + j];
  }

  // ---- self encoder: relu(self_feat @ W_se + b_se), K=24 padded to 32 ----
  {
    v16h a  = load_afrag_g(p.self_feat, 24, b0, 24, lm, hi);
    v16h bf = load_bfrag_g(p.W_se, 64, 0, cb, 24, lm, hi);
    v8f c = {};
    c = wmma16(a, bf, c);
    float bias = p.b_se[cb + lm];
#pragma unroll
    for (int r = 0; r < 8; ++r)
      comb_h[r + 8 * hi][cb + lm] = (_Float16)fmaxf(c[r] + bias, 0.f);
  }
  __syncthreads();

  // ---- q = self_embed @ W_q + b_q ----
  {
    v8f c = {};
#pragma unroll
    for (int s = 0; s < 2; ++s) {
      v16h a  = load_afrag_lds(&comb_h[0][0], 128, 0, s * 32, lm, hi);
      v16h bf = load_bfrag_g(p.W_q, 64, s * 32, cb, 64, lm, hi);
      c = wmma16(a, bf, c);
    }
    float bias = p.b_q[cb + lm];
#pragma unroll
    for (int r = 0; r < 8; ++r)
      q_s[r + 8 * hi][cb + lm] = c[r] + bias;
  }

  // ---- resident weight fragments for the entity pipeline ----
  v16h ee_f  = load_bfrag_g(p.W_ee, 64,  0, cb, 12, lm, hi);
  v16h wk_f0 = load_bfrag_g(p.W_k,  64,  0, cb, 64, lm, hi);
  v16h wk_f1 = load_bfrag_g(p.W_k,  64, 32, cb, 64, lm, hi);
  v16h wv_f0 = load_bfrag_g(p.W_v,  64,  0, cb, 64, lm, hi);
  v16h wv_f1 = load_bfrag_g(p.W_v,  64, 32, cb, 64, lm, hi);
  const float bk = p.b_k[cb + lm];
  const float bv = p.b_v[cb + lm];
  const float be = p.b_ee[cb + lm];

  // encode batch row (b0+row) -> ent_h[buf]
  auto encode_row = [&](int row, int buf) {
    (void)row;
    for (int g = 0; g < 8; ++g) {
#if HAVE_TDM
      v16h a = load_afrag_lds_f32(&fbuf[buf][0], 12, g * 16, 12, lm, hi);
#else
      v16h a = load_afrag_g(p.entities, 12, (long long)(b0 + row) * 128 + g * 16, 12, lm, hi);
#endif
      v8f c = {};
      c = wmma16(a, ee_f, c);
#pragma unroll
      for (int r = 0; r < 8; ++r)
        ent_h[buf][g * 16 + r + 8 * hi][cb + lm] = (_Float16)fmaxf(c[r] + be, 0.f);
    }
  };

  // masked online-softmax attention for head wv over ent_h[buf]
  auto attend_row = [&](int row, int buf) {
    float m_run = NEG_INF_F, s_run = 0.f, acc = 0.f;
    const float qd = q_s[row][cb + lm];       // q[row, head=wv, dim=lm]
    for (int g = 0; g < 8; ++g) {
      v16h a0 = load_afrag_lds(&ent_h[buf][0][0], 64, g * 16,  0, lm, hi);
      v16h a1 = load_afrag_lds(&ent_h[buf][0][0], 64, g * 16, 32, lm, hi);
      v8f kt = {}, vt = {};
      kt = wmma16(a0, wk_f0, kt); kt = wmma16(a1, wk_f1, kt);
      vt = wmma16(a0, wv_f0, vt); vt = wmma16(a1, wv_f1, vt);

      unsigned int mw = mbits[row][g >> 1];
      int bitbase = (g & 1) * 16 + 8 * hi;    // this half's 8 entities
      float sc[8], vv[8];
      float gmax_h = NEG_INF_F;
#pragma unroll
      for (int r = 0; r < 8; ++r) {
        float part = (kt[r] + bk) * qd;       // per-dim partial of K.q
        part += __shfl_xor(part, 1, 32);
        part += __shfl_xor(part, 2, 32);
        part += __shfl_xor(part, 4, 32);
        part += __shfl_xor(part, 8, 32);      // sum over 16 dims within half
        float s = part * 0.25f;               // / sqrt(16)
        s = ((mw >> (bitbase + r)) & 1u) ? s : NEG_INF_F;
        sc[r] = s;
        gmax_h = fmaxf(gmax_h, s);
        vv[r] = vt[r] + bv;
      }
      float gmax = fmaxf(gmax_h, __shfl_xor(gmax_h, 16, 32));
      float mn = fmaxf(m_run, gmax);
      float sf = __expf(m_run - mn);
      float ph = 0.f, av = 0.f;
#pragma unroll
      for (int r = 0; r < 8; ++r) {
        float pe = (sc[r] <= -1e29f) ? 0.f : __expf(sc[r] - mn);
        ph += pe;
        av += pe * vv[r];                     // this half's 8 entities
      }
      float psum = ph + __shfl_xor(ph, 16, 32);
      s_run = s_run * sf + psum;
      acc   = acc   * sf + av;
      m_run = mn;
    }
    float tot  = acc + __shfl_xor(acc, 16, 32);
    float attn = (s_run > 0.f) ? tot / s_run : 0.f;  // nan_to_num
    if (hi == 0) comb_h[row][64 + cb + lm] = (_Float16)attn;
  };

  // ---- entity pipeline: TDM prefetch 2 ahead, encode 1 ahead, 1 barrier/row ----
#if HAVE_TDM
  if (wv == 0) {
    tdm_load_1d(p.entities + (long long)b0 * 1536,
                (unsigned)(unsigned long long)(const void*)&fbuf[0][0], 1536);
    __builtin_amdgcn_s_wait_tensorcnt(0);            // row 0 features arrived
    tdm_load_1d(p.entities + (long long)(b0 + 1) * 1536,
                (unsigned)(unsigned long long)(const void*)&fbuf[1][0], 1536);
  }
#endif
  __syncthreads();
  encode_row(0, 0);
  for (int b = 0; b < 16; ++b) {
#if HAVE_TDM
    __builtin_amdgcn_s_wait_tensorcnt(0);            // feat(b+1) ready (wave0; others trivial)
#endif
    __syncthreads();                                  // ent_h[b&1] ready; buffers recyclable
#if HAVE_TDM
    if (wv == 0 && b + 2 < 16)
      tdm_load_1d(p.entities + (long long)(b0 + b + 2) * 1536,
                  (unsigned)(unsigned long long)(const void*)&fbuf[b & 1][0], 1536);
#endif
    if (b + 1 < 16) encode_row(b + 1, (b + 1) & 1);
    attend_row(b, b & 1);
  }
  __syncthreads();

  // ---- r1 = relu(combined @ W_p1 + b_p1) ----
  {
    v16h af[4];
#pragma unroll
    for (int s = 0; s < 4; ++s)
      af[s] = load_afrag_lds(&comb_h[0][0], 128, 0, s * 32, lm, hi);
    for (int cg = 0; cg < 2; ++cg) {
      int c2 = wv * 32 + cg * 16;
      v8f c = {};
#pragma unroll
      for (int s = 0; s < 4; ++s) {
        v16h bf = load_bfrag_g(p.W_p1, 128, s * 32, c2, 128, lm, hi);
        c = wmma16(af[s], bf, c);
      }
      float bias = p.b_p1[c2 + lm];
#pragma unroll
      for (int r = 0; r < 8; ++r)
        r1_h[r + 8 * hi][c2 + lm] = (_Float16)fmaxf(c[r] + bias, 0.f);
    }
  }
  __syncthreads();

  // ---- reactive = relu(r1 @ W_p2 + b_p2) ----
  {
    v16h af[4];
#pragma unroll
    for (int s = 0; s < 4; ++s)
      af[s] = load_afrag_lds(&r1_h[0][0], 128, 0, s * 32, lm, hi);
    for (int cg = 0; cg < 2; ++cg) {
      int c2 = wv * 32 + cg * 16;
      v8f c = {};
#pragma unroll
      for (int s = 0; s < 4; ++s) {
        v16h bf = load_bfrag_g(p.W_p2, 128, s * 32, c2, 128, lm, hi);
        c = wmma16(af[s], bf, c);
      }
      float bias = p.b_p2[c2 + lm];
#pragma unroll
      for (int r = 0; r < 8; ++r)
        react_h[r + 8 * hi][c2 + lm] = (_Float16)fmaxf(c[r] + bias, 0.f);
    }
  }
  __syncthreads();

  // ---- LSTM: gates = reactive @ W_ih + hx @ W_hh + b; elementwise in-register ----
  for (int cg = 0; cg < 2; ++cg) {
    int cbL = wv * 32 + cg * 16;
    v8f zero = {};
    v8f gacc[4] = {zero, zero, zero, zero};
#pragma unroll
    for (int s = 0; s < 4; ++s) {
      v16h a1 = load_afrag_lds(&react_h[0][0], 128, 0, s * 32, lm, hi);
      v16h a2 = load_afrag_lds(&hx_h[0][0],    128, 0, s * 32, lm, hi);
#pragma unroll
      for (int gi = 0; gi < 4; ++gi) {   // i, f, g, o (PyTorch order)
        v16h bf1 = load_bfrag_g(p.W_ih, 512, s * 32, gi * 128 + cbL, 128, lm, hi);
        gacc[gi] = wmma16(a1, bf1, gacc[gi]);
        v16h bf2 = load_bfrag_g(p.W_hh, 512, s * 32, gi * 128 + cbL, 128, lm, hi);
        gacc[gi] = wmma16(a2, bf2, gacc[gi]);
      }
    }
    float bI = p.b_ih[0 * 128 + cbL + lm] + p.b_hh[0 * 128 + cbL + lm];
    float bF = p.b_ih[1 * 128 + cbL + lm] + p.b_hh[1 * 128 + cbL + lm];
    float bG = p.b_ih[2 * 128 + cbL + lm] + p.b_hh[2 * 128 + cbL + lm];
    float bO = p.b_ih[3 * 128 + cbL + lm] + p.b_hh[3 * 128 + cbL + lm];
#pragma unroll
    for (int r = 0; r < 8; ++r) {
      int b = r + 8 * hi;
      int j = cbL + lm;
      float ig = sigm(gacc[0][r] + bI);
      float fg = sigm(gacc[1][r] + bF);
      float gg = tanhf(gacc[2][r] + bG);
      float og = sigm(gacc[3][r] + bO);
      float cxv = p.cx[(long long)(b0 + b) * 128 + j];
      float cxn = fg * cxv + ig * gg;
      float hxn = og * tanhf(cxn);
      long long ob = (long long)(b0 + b) * OUT_COLS;
      p.out[ob + 12 + j]  = hxn;   // hx_new
      p.out[ob + 140 + j] = cxn;   // cx_new
      feat_h[b][j] = (_Float16)((float)react_h[b][j] + hxn);
    }
  }
  __syncthreads();

  // ---- h = relu(features @ W_a + b_a) ----
  {
    v16h af[4];
#pragma unroll
    for (int s = 0; s < 4; ++s)
      af[s] = load_afrag_lds(&feat_h[0][0], 128, 0, s * 32, lm, hi);
    for (int cg = 0; cg < 2; ++cg) {
      int c2 = wv * 32 + cg * 16;
      v8f c = {};
#pragma unroll
      for (int s = 0; s < 4; ++s) {
        v16h bf = load_bfrag_g(p.W_a, 128, s * 32, c2, 128, lm, hi);
        c = wmma16(af[s], bf, c);
      }
      float bias = p.b_a[c2 + lm];
#pragma unroll
      for (int r = 0; r < 8; ++r)
        hh_h[r + 8 * hi][c2 + lm] = (_Float16)fmaxf(c[r] + bias, 0.f);
    }
  }
  __syncthreads();

  // ---- tiny actor heads (12 outputs/row) as VALU dots ----
  for (int t = tid; t < 16 * 12; t += 128) {
    int b = t / 12, c = t % 12;
    const float* Wp; const float* bp; int stride, ci;
    if      (c < 3)  { Wp = p.W_mx; bp = p.b_mx; stride = 3; ci = c;     }
    else if (c < 6)  { Wp = p.W_my; bp = p.b_my; stride = 3; ci = c - 3; }
    else if (c < 8)  { Wp = p.W_fi; bp = p.b_fi; stride = 2; ci = c - 6; }
    else if (c < 10) { Wp = p.W_he; bp = p.b_he; stride = 2; ci = c - 8; }
    else if (c ==10) { Wp = p.W_ra; bp = p.b_ra; stride = 1; ci = 0;     }
    else             { Wp = p.W_rb; bp = p.b_rb; stride = 1; ci = 0;     }
    float acc = bp[ci];
    for (int j = 0; j < 128; ++j)
      acc += (float)hh_h[b][j] * Wp[j * stride + ci];
    if (c >= 10) {
      float sp = (acc > 20.f) ? acc : log1pf(__expf(acc));
      acc = sp + 1.f;
    }
    p.out[(long long)(b0 + b) * OUT_COLS + c] = acc;
  }
}

extern "C" void kernel_launch(void* const* d_in, const int* in_sizes, int n_in,
                              void* d_out, int out_size, void* d_ws, size_t ws_size,
                              hipStream_t stream) {
  (void)n_in; (void)d_ws; (void)ws_size; (void)out_size;
  Params p;
  p.self_feat = (const float*)d_in[0];
  p.entities  = (const float*)d_in[1];
  p.mask      = (const int*)  d_in[2];
  p.hx        = (const float*)d_in[3];
  p.cx        = (const float*)d_in[4];
  p.W_se = (const float*)d_in[5];  p.b_se = (const float*)d_in[6];
  p.W_ee = (const float*)d_in[7];  p.b_ee = (const float*)d_in[8];
  p.W_q  = (const float*)d_in[9];  p.b_q  = (const float*)d_in[10];
  p.W_k  = (const float*)d_in[11]; p.b_k  = (const float*)d_in[12];
  p.W_v  = (const float*)d_in[13]; p.b_v  = (const float*)d_in[14];
  p.W_p1 = (const float*)d_in[15]; p.b_p1 = (const float*)d_in[16];
  p.W_p2 = (const float*)d_in[17]; p.b_p2 = (const float*)d_in[18];
  p.W_ih = (const float*)d_in[19]; p.b_ih = (const float*)d_in[20];
  p.W_hh = (const float*)d_in[21]; p.b_hh = (const float*)d_in[22];
  p.W_a  = (const float*)d_in[23]; p.b_a  = (const float*)d_in[24];
  p.W_mx = (const float*)d_in[25]; p.b_mx = (const float*)d_in[26];
  p.W_my = (const float*)d_in[27]; p.b_my = (const float*)d_in[28];
  p.W_fi = (const float*)d_in[29]; p.b_fi = (const float*)d_in[30];
  p.W_he = (const float*)d_in[31]; p.b_he = (const float*)d_in[32];
  p.W_ra = (const float*)d_in[33]; p.b_ra = (const float*)d_in[34];
  p.W_rb = (const float*)d_in[35]; p.b_rb = (const float*)d_in[36];
  p.out  = (float*)d_out;

  const int B = in_sizes[0] / 24;     // SELF_DIM = 24
  const int grid = B / 16;
  aac_fused_kernel<<<grid, 128, 0, stream>>>(p);
}
